// Model_56745107915018
// MI455X (gfx1250) — matrix operand order repaired and checked by reading, not compiled, not run on hardware
//
#include <hip/hip_runtime.h>
#include <math.h>

typedef _Float16 half_t;
typedef __attribute__((ext_vector_type(16))) _Float16 v16h;
typedef __attribute__((ext_vector_type(8)))  _Float16 v8h;
typedef __attribute__((ext_vector_type(8)))  float    v8f;

#define B_      256
#define SEQ_    1024
#define NP      127
#define D_      512
#define H_      8
#define DH_     64
#define FF_     2048
#define LAYERS_ 8
#define PRED_   96
#define QKV_    512
#define NTOK    (B_ * NP)   // 32512 = 127 * 256

#define USE_ASYNC 1

union AFrag { v16h v; v8h h[2]; };

__device__ inline v8f wmma16(v16h a, v16h b, v8f c) {
  // D = A(16x32 f16) * B(32x16 f16) + C(16x16 f32)
  return __builtin_amdgcn_wmma_f32_16x16x32_f16(false, a, false, b, (short)0, c, false, false);
}

__device__ inline void vzero8(v8f& a) {
#pragma unroll
  for (int i = 0; i < 8; ++i) a[i] = 0.f;
}

// 16-byte tile-staging copy: CDNA5 async global->LDS (ASYNCcnt-tracked DMA),
// with a synchronous fallback.
__device__ inline void copy16(half_t* lds, const half_t* g) {
#if USE_ASYNC
  asm volatile("global_load_async_to_lds_b128 %0, %1, off"
               :: "v"((unsigned)(size_t)lds), "v"((unsigned long long)(size_t)g)
               : "memory");
#else
  *(uint4*)lds = *(const uint4*)g;
#endif
}
__device__ inline void copy_fence() {
#if USE_ASYNC
  asm volatile("s_wait_asynccnt 0x0" ::: "memory");
#endif
}

// ---------------------------------------------------------------------------
// Generic f16 GEMM: out = epilogue(A[MxK] * W[KxN] + bias), W given TRANSPOSED
// as WT[N][K]. Tile: 256(M) x 64(N), K step 32, double-buffered async staging.
// 256 threads = 8 waves; wave w owns rows [w*32, w*32+32) = 2 A-frags x 4 B.
// EPI 0: store f16.  EPI 1: tanh-GELU, store f16.  EPI 2: res + acc, store f32.
// ---------------------------------------------------------------------------
template<int EPI>
__global__ __launch_bounds__(256) void gemm_kernel(
    const half_t* __restrict__ A, const half_t* __restrict__ WT,
    const float* __restrict__ bias, const float* __restrict__ res,
    void* __restrict__ outp, int M, int N, int K)
{
  __shared__ half_t sA[2][256 * 40];  // row-major, pad 40 halves (80B)
  __shared__ half_t sB[2][64 * 40];   // WT rows: sB[n][k]
  const int m0 = blockIdx.x * 256;
  const int n0 = blockIdx.y * 64;
  const int tid = threadIdx.x;
  const int wave = tid >> 5, lane = tid & 31;
  const int ml = lane & 15, hs = lane >> 4;

  auto issue_tile = [&](int buf, int k0) {
    // A tile: 256 rows x 32 halves = 1024 x 16B chunks
#pragma unroll
    for (int r = 0; r < 4; ++r) {
      const int idx = tid + 256 * r;
      const int row = idx >> 2, cc = idx & 3;
      copy16(&sA[buf][row * 40 + cc * 8],
             A + (size_t)(m0 + row) * K + k0 + cc * 8);
    }
    // B tile: 64 rows (n) x 32 halves (k) = 256 x 16B chunks
    {
      const int row = tid >> 2, cc = tid & 3;
      copy16(&sB[buf][row * 40 + cc * 8],
             WT + (size_t)(n0 + row) * K + k0 + cc * 8);
    }
  };

  v8f acc0[4], acc1[4];
#pragma unroll
  for (int i = 0; i < 4; ++i) { vzero8(acc0[i]); vzero8(acc1[i]); }

  const int nk = K >> 5;
  issue_tile(0, 0);
  copy_fence();
  __syncthreads();

  for (int t = 0; t < nk; ++t) {
    const int cur = t & 1, nxt = cur ^ 1;
    if (t + 1 < nk) issue_tile(nxt, (t + 1) << 5);

    AFrag a0, a1;
    const half_t* ap = sA[cur] + (wave * 32 + ml) * 40 + hs * 8;
    a0.h[0] = *(const v8h*)ap;
    a0.h[1] = *(const v8h*)(ap + 16);
    const half_t* ap1 = ap + 16 * 40;
    a1.h[0] = *(const v8h*)ap1;
    a1.h[1] = *(const v8h*)(ap1 + 16);
#pragma unroll
    for (int nt = 0; nt < 4; ++nt) {
      AFrag b;
      const half_t* bp = sB[cur] + (nt * 16 + ml) * 40 + hs * 8;
      b.h[0] = *(const v8h*)bp;
      b.h[1] = *(const v8h*)(bp + 16);
      acc0[nt] = wmma16(a0.v, b.v, acc0[nt]);
      acc1[nt] = wmma16(a1.v, b.v, acc1[nt]);
    }
    copy_fence();
    __syncthreads();
  }

  // epilogue; C layout: row = v + 8*hs, col = ml
  const int rowb = m0 + wave * 32 + hs * 8;
#pragma unroll
  for (int nt = 0; nt < 4; ++nt) {
    const int col = n0 + nt * 16 + ml;
    const float bz = bias[col];
#pragma unroll
    for (int v = 0; v < 8; ++v) {
#pragma unroll
      for (int half_sel = 0; half_sel < 2; ++half_sel) {
        const int row = rowb + half_sel * 16 + v;
        const size_t o = (size_t)row * N + col;
        float xv = (half_sel ? acc1[nt][v] : acc0[nt][v]) + bz;
        if (EPI == 1) {
          const float tq = 0.7978845608f * (xv + 0.044715f * xv * xv * xv);
          xv = 0.5f * xv * (1.f + tanhf(tq));
        }
        if (EPI == 2) ((float*)outp)[o] = res[o] + xv;
        else          ((half_t*)outp)[o] = (half_t)xv;
      }
    }
  }
}

// ---------------------------------------------------------------------------
// Attention: one block per (b, head). 127 positions, DH=64.
// ---------------------------------------------------------------------------
__global__ __launch_bounds__(256) void attention_kernel(
    const half_t* __restrict__ qkv, half_t* __restrict__ attout)
{
  __shared__ half_t sK[128 * 72];    // [pos][dh] — transposed-B source for Q*K^T
  __shared__ half_t sVT[64 * 136];   // [dh][pos] — transposed-B source for P*V
  __shared__ half_t sP[8][16 * 40];  // per-wave P chunk bounce (16 x 32)

  const int bh = blockIdx.x;
  const int b = bh >> 3, h = bh & 7;
  const size_t tokbase = (size_t)b * NP;
  const int tid = threadIdx.x;
  const int wave = tid >> 5, lane = tid & 31;
  const int ml = lane & 15, hs = lane >> 4;

  for (int idx = tid; idx < 1024; idx += 256) {
    const int p = idx >> 3, c = idx & 7;
    uint4 kv = make_uint4(0u, 0u, 0u, 0u);
    uint4 vv = make_uint4(0u, 0u, 0u, 0u);
    if (p < NP) {
      const half_t* base = qkv + (tokbase + p) * (3 * QKV_) + h * DH_ + c * 8;
      kv = *(const uint4*)(base + QKV_);
      vv = *(const uint4*)(base + 2 * QKV_);
    }
    *(uint4*)(sK + p * 72 + c * 8) = kv;
    const half_t* hv = (const half_t*)&vv;
#pragma unroll
    for (int j = 0; j < 8; ++j) sVT[(c * 8 + j) * 136 + p] = hv[j];
  }
  __syncthreads();

  const int qbase = wave * 16;
  int qrow = qbase + ml; if (qrow > NP - 1) qrow = NP - 1;
  const half_t* qp = qkv + (tokbase + qrow) * (3 * QKV_) + h * DH_;
  AFrag aq[2];
#pragma unroll
  for (int kc = 0; kc < 2; ++kc) {
    const half_t* p0 = qp + kc * 32 + hs * 8;
    aq[kc].h[0] = *(const v8h*)p0;
    aq[kc].h[1] = *(const v8h*)(p0 + 16);
  }

  v8f S[8];
#pragma unroll
  for (int i = 0; i < 8; ++i) vzero8(S[i]);
#pragma unroll
  for (int ct = 0; ct < 8; ++ct) {
#pragma unroll
    for (int kc = 0; kc < 2; ++kc) {
      AFrag bk;
      const half_t* bp = sK + (ct * 16 + ml) * 72 + kc * 32 + hs * 8;
      bk.h[0] = *(const v8h*)bp;
      bk.h[1] = *(const v8h*)(bp + 16);
      S[ct] = wmma16(aq[kc].v, bk.v, S[ct]);
    }
  }

  float rsum[8];
#pragma unroll
  for (int v = 0; v < 8; ++v) {
    float mx = -1e30f;
#pragma unroll
    for (int ct = 0; ct < 8; ++ct) {
      const float s = S[ct][v] * 0.125f;  // DH^-0.5
      S[ct][v] = s;
      if (ct * 16 + ml < NP) mx = fmaxf(mx, s);
    }
#pragma unroll
    for (int off = 1; off < 16; off <<= 1) mx = fmaxf(mx, __shfl_xor(mx, off, 32));
    float sum = 0.f;
#pragma unroll
    for (int ct = 0; ct < 8; ++ct) {
      const float e = (ct * 16 + ml < NP) ? __expf(S[ct][v] - mx) : 0.f;
      S[ct][v] = e;
      sum += e;
    }
#pragma unroll
    for (int off = 1; off < 16; off <<= 1) sum += __shfl_xor(sum, off, 32);
    rsum[v] = sum;
  }

  v8f O[4];
#pragma unroll
  for (int i = 0; i < 4; ++i) vzero8(O[i]);
  for (int c = 0; c < 4; ++c) {
#pragma unroll
    for (int hf = 0; hf < 2; ++hf) {
      const int ct = 2 * c + hf;
#pragma unroll
      for (int v = 0; v < 8; ++v)
        sP[wave][(v + 8 * hs) * 40 + hf * 16 + ml] = (half_t)S[ct][v];
    }
    __syncthreads();
    AFrag pa;
    const half_t* pp = sP[wave] + ml * 40 + hs * 8;
    pa.h[0] = *(const v8h*)pp;
    pa.h[1] = *(const v8h*)(pp + 16);
#pragma unroll
    for (int nt = 0; nt < 4; ++nt) {
      AFrag bv;
      const half_t* bp = sVT + (nt * 16 + ml) * 136 + c * 32 + hs * 8;
      bv.h[0] = *(const v8h*)bp;
      bv.h[1] = *(const v8h*)(bp + 16);
      O[nt] = wmma16(pa.v, bv.v, O[nt]);
    }
    __syncthreads();
  }

#pragma unroll
  for (int nt = 0; nt < 4; ++nt) {
#pragma unroll
    for (int v = 0; v < 8; ++v) {
      const int row = qbase + v + 8 * hs;
      if (row < NP) {
        const float o = O[nt][v] / rsum[v];
        attout[(tokbase + row) * QKV_ + h * DH_ + nt * 16 + ml] = (half_t)o;
      }
    }
  }
}

// ---------------------------------------------------------------------------
__global__ __launch_bounds__(256) void rmsnorm_kernel(
    const float* __restrict__ x, const float* __restrict__ w, half_t* __restrict__ out)
{
  const size_t tok = blockIdx.x;
  const float* xr = x + tok * D_;
  const int tid = threadIdx.x;
  const float v0 = xr[tid], v1 = xr[tid + 256];
  float ss = v0 * v0 + v1 * v1;
#pragma unroll
  for (int off = 16; off > 0; off >>= 1) ss += __shfl_xor(ss, off, 32);
  __shared__ float red[8];
  if ((tid & 31) == 0) red[tid >> 5] = ss;
  __syncthreads();
  float tot = 0.f;
#pragma unroll
  for (int i = 0; i < 8; ++i) tot += red[i];
  const float r = rsqrtf(tot * (1.f / 512.f) + 1e-6f);
  out[tok * D_ + tid]       = (half_t)(v0 * r * (1.f + w[tid]));
  out[tok * D_ + tid + 256] = (half_t)(v1 * r * (1.f + w[tid + 256]));
}

__global__ __launch_bounds__(256) void input_norm_kernel(
    const float* __restrict__ x, float* __restrict__ xn,
    float* __restrict__ means, float* __restrict__ stdev)
{
  const int b = blockIdx.x;
  const float* xr = x + (size_t)b * SEQ_;
  const int tid = threadIdx.x;
  float vals[4]; float s = 0.f, s2 = 0.f;
#pragma unroll
  for (int i = 0; i < 4; ++i) { const float v = xr[tid + 256 * i]; vals[i] = v; s += v; s2 += v * v; }
#pragma unroll
  for (int off = 16; off > 0; off >>= 1) { s += __shfl_xor(s, off, 32); s2 += __shfl_xor(s2, off, 32); }
  __shared__ float r1[8], r2[8];
  if ((tid & 31) == 0) { r1[tid >> 5] = s; r2[tid >> 5] = s2; }
  __syncthreads();
  float ts = 0.f, ts2 = 0.f;
#pragma unroll
  for (int i = 0; i < 8; ++i) { ts += r1[i]; ts2 += r2[i]; }
  const float mean = ts * (1.f / SEQ_);
  const float var = ts2 * (1.f / SEQ_) - mean * mean;
  const float sd = sqrtf(var + 1e-5f);
  if (tid == 0) { means[b] = mean; stdev[b] = sd; }
  const float inv = 1.f / sd;
#pragma unroll
  for (int i = 0; i < 4; ++i) xn[(size_t)b * SEQ_ + tid + 256 * i] = (vals[i] - mean) * inv;
}

__global__ __launch_bounds__(256) void patch_embed_kernel(
    const float* __restrict__ xn, const float* __restrict__ We,
    const float* __restrict__ be, float* __restrict__ h)
{
  const int tokp = blockIdx.x;
  const int b = tokp / NP, pp = tokp % NP;
  const float* src = xn + (size_t)b * SEQ_ + pp * 8;
  __shared__ float patch[16];
  if (threadIdx.x < 16) patch[threadIdx.x] = src[threadIdx.x];
  __syncthreads();
  const int d = threadIdx.x;
  float a0 = be[d], a1 = be[d + 256];
#pragma unroll
  for (int l = 0; l < 16; ++l) {
    const float pv = patch[l];
    a0 += pv * We[l * D_ + d];
    a1 += pv * We[l * D_ + d + 256];
  }
  h[(size_t)tokp * D_ + d]       = a0;
  h[(size_t)tokp * D_ + d + 256] = a1;
}

// Fused transpose + f32->f16: WT[n][k] = (f16)W[k][n], per-layer via blockIdx.z.
__global__ __launch_bounds__(256) void transpose_f32_to_f16_kernel(
    const float* __restrict__ W, half_t* __restrict__ WT, int K, int N)
{
  const size_t moff = (size_t)blockIdx.z * K * N;
  __shared__ float tile[32][33];
  const int k0 = blockIdx.x * 32, n0 = blockIdx.y * 32;
  const int r = threadIdx.x >> 5, c = threadIdx.x & 31;
#pragma unroll
  for (int i = 0; i < 4; ++i)
    tile[r + 8 * i][c] = W[moff + (size_t)(k0 + r + 8 * i) * N + n0 + c];
  __syncthreads();
#pragma unroll
  for (int i = 0; i < 4; ++i)
    WT[moff + (size_t)(n0 + r + 8 * i) * K + k0 + c] = (half_t)tile[c][r + 8 * i];
}

// Head GEMM: M=256, N=96, K=65024; WT layout [96][65024]; K split over 127
// chunks of 512, f32 atomics into accumulator.
__global__ __launch_bounds__(256) void head_gemm_kernel(
    const half_t* __restrict__ A, const half_t* __restrict__ WT, float* __restrict__ accout)
{
  __shared__ half_t sA[128 * 40];
  __shared__ half_t sB[96 * 40];
  const int m0 = blockIdx.x * 128;
  const size_t kbase = (size_t)blockIdx.y * 512;
  const int KTOT = NP * D_;
  const int tid = threadIdx.x;
  const int wave = tid >> 5, lane = tid & 31;
  const int ml = lane & 15, hs = lane >> 4;

  v8f acc[6];
#pragma unroll
  for (int i = 0; i < 6; ++i) vzero8(acc[i]);

  for (int kk = 0; kk < 512; kk += 32) {
    {
      int idx = tid;
#pragma unroll
      for (int r = 0; r < 2; ++r, idx += 256) {
        const int row = idx >> 2, cc = idx & 3;
        *(uint4*)(sA + row * 40 + cc * 8) =
            *(const uint4*)(A + (size_t)(m0 + row) * KTOT + kbase + kk + cc * 8);
      }
    }
    for (int idx2 = tid; idx2 < 384; idx2 += 256) {
      const int n = idx2 >> 2, cc = idx2 & 3;
      *(uint4*)(sB + n * 40 + cc * 8) =
          *(const uint4*)(WT + (size_t)n * KTOT + kbase + kk + cc * 8);
    }
    __syncthreads();
    AFrag a;
    const half_t* ap = sA + (wave * 16 + ml) * 40 + hs * 8;
    a.h[0] = *(const v8h*)ap;
    a.h[1] = *(const v8h*)(ap + 16);
#pragma unroll
    for (int nt = 0; nt < 6; ++nt) {
      AFrag bfr;
      const half_t* bp = sB + (nt * 16 + ml) * 40 + hs * 8;
      bfr.h[0] = *(const v8h*)bp;
      bfr.h[1] = *(const v8h*)(bp + 16);
      acc[nt] = wmma16(a.v, bfr.v, acc[nt]);
    }
    __syncthreads();
  }
#pragma unroll
  for (int nt = 0; nt < 6; ++nt)
#pragma unroll
    for (int v = 0; v < 8; ++v)
      atomicAdd(&accout[(size_t)(m0 + wave * 16 + v + 8 * hs) * 96 + nt * 16 + ml], acc[nt][v]);
}

__global__ void mul_kernel(half_t* __restrict__ g, const half_t* __restrict__ u, size_t n) {
  size_t i = (size_t)blockIdx.x * blockDim.x + threadIdx.x;
  const size_t st = (size_t)gridDim.x * blockDim.x;
  for (; i < n; i += st) g[i] = (half_t)((float)g[i] * (float)u[i]);
}

__global__ void zero_kernel(float* __restrict__ p, int n) {
  const int i = blockIdx.x * 256 + threadIdx.x;
  if (i < n) p[i] = 0.f;
}

__global__ void head_final_kernel(const float* __restrict__ acc, const float* __restrict__ bh,
                                  const float* __restrict__ means, const float* __restrict__ stdev,
                                  float* __restrict__ out) {
  const int i = blockIdx.x * 256 + threadIdx.x;
  if (i < B_ * PRED_) {
    const int b = i / PRED_, j = i % PRED_;
    out[i] = (acc[i] + bh[j]) * stdev[b] + means[b];
  }
}

// ---------------------------------------------------------------------------
extern "C" void kernel_launch(void* const* d_in, const int* in_sizes, int n_in,
                              void* d_out, int out_size, void* d_ws, size_t ws_size,
                              hipStream_t stream)
{
  (void)in_sizes; (void)n_in; (void)out_size; (void)ws_size;
  const float* x      = (const float*)d_in[0];
  const float* We     = (const float*)d_in[1];
  const float* be     = (const float*)d_in[2];
  const float* Wqkv   = (const float*)d_in[3];
  const float* bqkv   = (const float*)d_in[4];
  const float* Wo     = (const float*)d_in[5];
  const float* bo     = (const float*)d_in[6];
  const float* Wg     = (const float*)d_in[7];
  const float* bg     = (const float*)d_in[8];
  const float* Wu     = (const float*)d_in[9];
  const float* bu     = (const float*)d_in[10];
  const float* Wd     = (const float*)d_in[11];
  const float* bd     = (const float*)d_in[12];
  const float* w_in   = (const float*)d_in[13];
  const float* w_post = (const float*)d_in[14];
  const float* norm_w = (const float*)d_in[15];
  const float* Wh     = (const float*)d_in[16];
  const float* bh     = (const float*)d_in[17];

  char* p = (char*)d_ws;
  auto alloc = [&](size_t bytes) -> void* {
    void* r = (void*)p; p += (bytes + 255) & ~(size_t)255; return r;
  };
  float*  means   = (float*)alloc((size_t)B_ * 4);
  float*  stdev   = (float*)alloc((size_t)B_ * 4);
  float*  xn      = (float*)alloc((size_t)B_ * SEQ_ * 4);
  half_t* WqkvT   = (half_t*)alloc((size_t)LAYERS_ * D_ * 3 * QKV_ * 2);
  half_t* WoT     = (half_t*)alloc((size_t)LAYERS_ * QKV_ * D_ * 2);
  half_t* WgT     = (half_t*)alloc((size_t)LAYERS_ * D_ * FF_ * 2);
  half_t* WuT     = (half_t*)alloc((size_t)LAYERS_ * D_ * FF_ * 2);
  half_t* WdT     = (half_t*)alloc((size_t)LAYERS_ * FF_ * D_ * 2);
  half_t* WhT     = (half_t*)alloc((size_t)NP * D_ * PRED_ * 2);
  float*  hbuf    = (float*)alloc((size_t)NTOK * D_ * 4);
  half_t* hn      = (half_t*)alloc((size_t)NTOK * D_ * 2);
  half_t* reg1    = (half_t*)alloc((size_t)NTOK * FF_ * 2);  // qkv then gate
  half_t* reg2    = (half_t*)alloc((size_t)NTOK * FF_ * 2);  // att then up
  float*  headAcc = (float*)alloc((size_t)B_ * PRED_ * 4);

  half_t* qkvbuf = reg1;
  half_t* gbuf   = reg1;
  half_t* attbuf = reg2;
  half_t* ubuf   = reg2;

  input_norm_kernel<<<B_, 256, 0, stream>>>(x, xn, means, stdev);
  transpose_f32_to_f16_kernel<<<dim3(D_ / 32, (3 * QKV_) / 32, LAYERS_), 256, 0, stream>>>(
      Wqkv, WqkvT, D_, 3 * QKV_);
  transpose_f32_to_f16_kernel<<<dim3(QKV_ / 32, D_ / 32, LAYERS_), 256, 0, stream>>>(
      Wo, WoT, QKV_, D_);
  transpose_f32_to_f16_kernel<<<dim3(D_ / 32, FF_ / 32, LAYERS_), 256, 0, stream>>>(
      Wg, WgT, D_, FF_);
  transpose_f32_to_f16_kernel<<<dim3(D_ / 32, FF_ / 32, LAYERS_), 256, 0, stream>>>(
      Wu, WuT, D_, FF_);
  transpose_f32_to_f16_kernel<<<dim3(FF_ / 32, D_ / 32, LAYERS_), 256, 0, stream>>>(
      Wd, WdT, FF_, D_);
  transpose_f32_to_f16_kernel<<<dim3((NP * D_) / 32, PRED_ / 32, 1), 256, 0, stream>>>(
      Wh, WhT, NP * D_, PRED_);
  patch_embed_kernel<<<NTOK, 256, 0, stream>>>(xn, We, be, hbuf);

  for (int L = 0; L < LAYERS_; ++L) {
    rmsnorm_kernel<<<NTOK, 256, 0, stream>>>(hbuf, w_in + L * D_, hn);
    gemm_kernel<0><<<dim3(NTOK / 256, (3 * QKV_) / 64), 256, 0, stream>>>(
        hn, WqkvT + (size_t)L * D_ * 3 * QKV_, bqkv + L * 3 * QKV_, nullptr, qkvbuf,
        NTOK, 3 * QKV_, D_);
    attention_kernel<<<B_ * H_, 256, 0, stream>>>(qkvbuf, attbuf);
    gemm_kernel<2><<<dim3(NTOK / 256, D_ / 64), 256, 0, stream>>>(
        attbuf, WoT + (size_t)L * QKV_ * D_, bo + L * D_, hbuf, hbuf, NTOK, D_, QKV_);
    rmsnorm_kernel<<<NTOK, 256, 0, stream>>>(hbuf, w_post + L * D_, hn);
    gemm_kernel<1><<<dim3(NTOK / 256, FF_ / 64), 256, 0, stream>>>(
        hn, WgT + (size_t)L * D_ * FF_, bg + L * FF_, nullptr, gbuf, NTOK, FF_, D_);
    gemm_kernel<0><<<dim3(NTOK / 256, FF_ / 64), 256, 0, stream>>>(
        hn, WuT + (size_t)L * D_ * FF_, bu + L * FF_, nullptr, ubuf, NTOK, FF_, D_);
    mul_kernel<<<8192, 256, 0, stream>>>(gbuf, ubuf, (size_t)NTOK * FF_);
    gemm_kernel<2><<<dim3(NTOK / 256, D_ / 64), 256, 0, stream>>>(
        gbuf, WdT + (size_t)L * FF_ * D_, bd + L * D_, hbuf, hbuf, NTOK, D_, FF_);
  }

  rmsnorm_kernel<<<NTOK, 256, 0, stream>>>(hbuf, norm_w, hn);
  zero_kernel<<<(B_ * PRED_ + 255) / 256, 256, 0, stream>>>(headAcc, B_ * PRED_);
  head_gemm_kernel<<<dim3(2, NP), 256, 0, stream>>>(hn, WhT, headAcc);
  head_final_kernel<<<(B_ * PRED_ + 255) / 256, 256, 0, stream>>>(
      headAcc, bh, means, stdev, (float*)d_out);
}
